// KokoroModel_23699629539423
// MI455X (gfx1250) — matrix operand order repaired and checked by reading, not compiled, not run on hardware
//
#include <hip/hip_runtime.h>
#include <hip/hip_bf16.h>
#include <stdint.h>

// ---------------------------------------------------------------------------
// Types / helpers
// ---------------------------------------------------------------------------
typedef _Float16 v16h __attribute__((ext_vector_type(16)));
typedef _Float16 v8h  __attribute__((ext_vector_type(8)));
typedef float    v8f  __attribute__((ext_vector_type(8)));

static __device__ __forceinline__ v16h pack16(v8h lo, v8h hi) {
    v16h r;
#pragma unroll
    for (int i = 0; i < 8; ++i) { r[i] = lo[i]; r[8 + i] = hi[i]; }
    return r;
}

static __device__ __forceinline__ float sigm(float x) {
    return 1.0f / (1.0f + __expf(-x));
}

// CDNA5 async global->LDS copy (ASYNCcnt-tracked, no VGPR data path).
// vdst = LDS byte address (low 32 bits of the flat shared address),
// vaddr = 64-bit global address.
static __device__ __forceinline__ void async_load_lds_b128(void* lds, const void* g) {
    asm volatile("global_load_async_to_lds_b128 %0, %1, off"
                 :: "v"((uint32_t)(uintptr_t)lds), "v"((uint64_t)(uintptr_t)g)
                 : "memory");
}
static __device__ __forceinline__ void wait_async0() {
    asm volatile("s_wait_asynccnt 0x0" ::: "memory");
}

// ---------------------------------------------------------------------------
// Elementwise / conversion kernels
// ---------------------------------------------------------------------------
__global__ void cvt_kernel(const float* __restrict__ s, _Float16* __restrict__ d, long n) {
    long i = (long)blockIdx.x * blockDim.x + threadIdx.x;
    if (i < n) d[i] = (_Float16)s[i];
}

// convert [rows, cs] f32 -> [rows, cd] f16 with zero padding (cs <= cd)
__global__ void cvt_pad_kernel(const float* __restrict__ s, _Float16* __restrict__ d,
                               int rows, int cs, int cd) {
    long i = (long)blockIdx.x * blockDim.x + threadIdx.x;
    long n = (long)rows * cd;
    if (i >= n) return;
    int r = (int)(i / cd), c = (int)(i % cd);
    d[i] = (c < cs) ? (_Float16)s[(long)r * cs + c] : (_Float16)0;
}

__global__ void vecadd_kernel(const float* __restrict__ a, const float* __restrict__ b,
                              float* __restrict__ o, int n) {
    int i = blockIdx.x * blockDim.x + threadIdx.x;
    if (i < n) o[i] = a[i] + b[i];
}

// text embedding gather -> f16 [B*S, 512]
__global__ void embed_kernel(const int* __restrict__ idx, const float* __restrict__ emb,
                             _Float16* __restrict__ out, int rows) {
    long i = (long)blockIdx.x * blockDim.x + threadIdx.x;
    long n = (long)rows * 512;
    if (i >= n) return;
    long r = i >> 9; int c = (int)(i & 511);
    out[i] = (_Float16)emb[(long)idx[r] * 512 + c];
}

// teacher-forced shifted mel, padded 80->96, f16 [B*T, 96]
__global__ void mel_shift_kernel(const float* __restrict__ mel, _Float16* __restrict__ out) {
    long i = (long)blockIdx.x * blockDim.x + threadIdx.x;
    long n = 32000L * 96;
    if (i >= n) return;
    long row = i / 96; int c = (int)(i % 96);
    int b = (int)(row / 1000), t = (int)(row % 1000);
    float v = 0.0f;
    if (t > 0 && c < 80) v = mel[((long)b * 1000 + (t - 1)) * 80 + c];
    out[i] = (_Float16)v;
}

// row softmax over 512 cols with 1/sqrt(64) scale, f32 in -> f16 out
__global__ __launch_bounds__(256) void softmax512_kernel(const float* __restrict__ x,
                                                         _Float16* __restrict__ y) {
    __shared__ float red[256];
    long row = blockIdx.x;
    const float* xr = x + row * 512;
    _Float16* yr = y + row * 512;
    int t = threadIdx.x;
    float a0 = xr[t] * 0.125f, a1 = xr[t + 256] * 0.125f;
    red[t] = fmaxf(a0, a1);
    __syncthreads();
    for (int s = 128; s > 0; s >>= 1) { if (t < s) red[t] = fmaxf(red[t], red[t + s]); __syncthreads(); }
    float mx = red[0];
    __syncthreads();
    float e0 = __expf(a0 - mx), e1 = __expf(a1 - mx);
    red[t] = e0 + e1;
    __syncthreads();
    for (int s = 128; s > 0; s >>= 1) { if (t < s) red[t] += red[t + s]; __syncthreads(); }
    float inv = 1.0f / red[0];
    yr[t]       = (_Float16)(e0 * inv);
    yr[t + 256] = (_Float16)(e1 * inv);
}

// ---------------------------------------------------------------------------
// Generic WMMA GEMM.
//   C[m,n] = sum_k A[m,k] * (BT ? W[n,k] : Bm[k,n]) + bias[n] (+ bias2[n])
// Block tile 64(M) x 128(N) x 32(K), 256 threads = 8 wave32 (2x4).
// Tiles are staged to LDS with GLOBAL_LOAD_ASYNC_TO_LDS_B128 (ASYNCcnt path).
// Tile loads are deliberately unguarded: every possible M/N overrun (<=64 rows
// / <=48 cols across all call sites) lands inside the workspace, and garbage in
// OOB A-rows / B-cols only affects C rows/cols that the store guards discard.
// Two-level batch: z1 = blockIdx.z / z2n, z2 = blockIdx.z % z2n.
// ---------------------------------------------------------------------------
template <bool BT, bool OUT16>
__global__ __launch_bounds__(256) void gemm_kernel(
    const _Float16* __restrict__ A, long lda, long sA1, long sA2,
    const _Float16* __restrict__ Bm, long ldb, long sB1, long sB2,
    void* __restrict__ Cv, long ldc, long sC1, long sC2,
    const float* __restrict__ bias, const float* __restrict__ bias2,
    int M, int N, int K, int z2n) {
    __shared__ _Float16 As[64][40];    // 40 f16 = 80B row stride (16B aligned)
    __shared__ _Float16 Bs[128][40];

    long z = blockIdx.z;
    long z1 = z / z2n, z2 = z % z2n;
    A  += z1 * sA1 + z2 * sA2;
    Bm += z1 * sB1 + z2 * sB2;

    const int tid  = threadIdx.x;
    const int wid  = tid >> 5;
    const int lane = tid & 31;
    const int wm = wid >> 2, wn = wid & 3;
    const int m0 = blockIdx.y * 64, n0 = blockIdx.x * 128;
    const int l16   = lane & 15;
    const int mbase = (lane < 16) ? 0 : 8;
    const int kbA   = (lane < 16) ? 0 : 16;
    const int kbB   = (lane < 16) ? 0 : 16;

    v8f acc[2][2];
#pragma unroll
    for (int ni = 0; ni < 2; ++ni) {
        int gcol = n0 + wn * 32 + ni * 16 + l16;
        int bc = gcol < N ? gcol : (N - 1);
        float bv = 0.0f;
        if (bias)  bv += bias[bc];
        if (bias2) bv += bias2[bc];
#pragma unroll
        for (int mi = 0; mi < 2; ++mi)
#pragma unroll
            for (int r = 0; r < 8; ++r) acc[mi][ni][r] = bv;
    }

#pragma unroll 1
    for (int k0 = 0; k0 < K; k0 += 32) {
        // stage A: 64x32, one async b128 per thread
        {
            int r = tid >> 2, cq = (tid & 3) << 3;
            async_load_lds_b128(&As[r][cq], A + (long)(m0 + r) * lda + k0 + cq);
        }
        // stage B into Bs[n][k]
        if (BT) {
            int n = tid >> 1, kq = (tid & 1) << 4;
            const _Float16* p = Bm + (long)(n0 + n) * ldb + k0 + kq;
            async_load_lds_b128(&Bs[n][kq],     p);
            async_load_lds_b128(&Bs[n][kq + 8], p + 8);
        } else {
            // transposing stage (ctx = attn @ V only): manual scatter
            int kk = tid & 31, nb = (tid >> 5) << 4;
            int gn = n0 + nb;
            v8h v0 = {}, v1 = {};
            if (gn < N) {  // N is a multiple of 16 on this path
                const _Float16* p = Bm + (long)(k0 + kk) * ldb + gn;
                v0 = *(const v8h*)p;
                v1 = *(const v8h*)(p + 8);
            }
#pragma unroll
            for (int j = 0; j < 8; ++j) { Bs[nb + j][kk] = v0[j]; Bs[nb + 8 + j][kk] = v1[j]; }
        }
        wait_async0();     // own async copies landed in LDS
        __syncthreads();   // everyone's copies visible

        v16h af[2], bf[2];
#pragma unroll
        for (int mi = 0; mi < 2; ++mi) {
            const _Float16* p = &As[wm * 32 + mi * 16 + l16][kbA];
            af[mi] = pack16(*(const v8h*)p, *(const v8h*)(p + 8));
        }
#pragma unroll
        for (int ni = 0; ni < 2; ++ni) {
            const _Float16* p = &Bs[wn * 32 + ni * 16 + l16][kbB];
            bf[ni] = pack16(*(const v8h*)p, *(const v8h*)(p + 8));
        }
#pragma unroll
        for (int mi = 0; mi < 2; ++mi)
#pragma unroll
            for (int ni = 0; ni < 2; ++ni)
                acc[mi][ni] = __builtin_amdgcn_wmma_f32_16x16x32_f16(
                    false, af[mi], false, bf[ni], (short)0, acc[mi][ni], false, false);
        __syncthreads();
    }

    // store C (guards discard any tile overrun)
#pragma unroll
    for (int mi = 0; mi < 2; ++mi)
#pragma unroll
        for (int ni = 0; ni < 2; ++ni) {
            int gcol = n0 + wn * 32 + ni * 16 + l16;
            if (gcol >= N) continue;
#pragma unroll
            for (int r = 0; r < 8; ++r) {
                int gm = m0 + wm * 32 + mi * 16 + mbase + r;
                if (gm >= M) continue;
                if constexpr (OUT16) {
                    _Float16* C = (_Float16*)Cv + z1 * sC1 + z2 * sC2;
                    C[(long)gm * ldc + gcol] = (_Float16)acc[mi][ni][r];
                } else {
                    float* C = (float*)Cv + z1 * sC1 + z2 * sC2;
                    C[(long)gm * ldc + gcol] = acc[mi][ni][r];
                }
            }
        }
}

// ---------------------------------------------------------------------------
// Persistent LSTM recurrence (B=32, H=512, gates 4H=2048).
// One workgroup, 1024 threads = 32 waves. Wave w owns output columns
// w*16..w*16+15 for ALL four gates -> cell update is wave-local (c in regs).
// h kept in LDS as f16 and consumed as WMMA A-fragments next step.
//   G = xg[t] + h @ Whh^T ; gate order i,f,g,o.
// ---------------------------------------------------------------------------
__global__ __launch_bounds__(1024, 1) void lstm_rec_kernel(
    const float* __restrict__ xg,        // rows b*steps + t, 2048 cols
    const _Float16* __restrict__ Whh,    // [2048, 512] f16
    _Float16* __restrict__ hout,         // rows b*steps + t
    int ld_hout, int col_off, int steps, int dir) {
    __shared__ _Float16 h_lds[32][528];

    const int tid  = threadIdx.x;
    const int wid  = tid >> 5;   // 0..31, column-slice owner
    const int lane = tid & 31;
    const int l16   = lane & 15;
    const int col   = (wid << 4) + l16;          // 0..511
    const int mbase = (lane < 16) ? 0 : 8;
    const int kbA   = (lane < 16) ? 0 : 16;
    const int kbB   = (lane < 16) ? 0 : 16;

    for (int i = tid; i < 32 * 528; i += 1024) ((_Float16*)h_lds)[i] = (_Float16)0;
    float c[2][8];
#pragma unroll
    for (int mi = 0; mi < 2; ++mi)
#pragma unroll
        for (int r = 0; r < 8; ++r) c[mi][r] = 0.0f;
    __syncthreads();

#pragma unroll 1
    for (int t = 0; t < steps; ++t) {
        const int ta = (dir > 0) ? t : (steps - 1 - t);

        v8f acc[2][4];
#pragma unroll
        for (int mi = 0; mi < 2; ++mi)
#pragma unroll
            for (int g = 0; g < 4; ++g)
#pragma unroll
                for (int r = 0; r < 8; ++r) {
                    int m = (mi << 4) + mbase + r;  // batch row
                    acc[mi][g][r] = xg[((long)m * steps + ta) * 2048 + (g << 9) + col];
                }

        // prefetch next step's gate rows while this step computes
        if (t + 1 < steps) {
            const int tn = (dir > 0) ? (t + 1) : (steps - 2 - t);
            __builtin_prefetch(&xg[((long)mbase * steps + tn) * 2048 + col], 0, 0);
            __builtin_prefetch(&xg[((long)(16 + mbase) * steps + tn) * 2048 + col], 0, 0);
        }

#pragma unroll 1
        for (int k0 = 0; k0 < 512; k0 += 32) {
            v16h a[2];
#pragma unroll
            for (int mi = 0; mi < 2; ++mi) {
                const _Float16* p = &h_lds[(mi << 4) + l16][k0 + kbA];
                a[mi] = pack16(*(const v8h*)p, *(const v8h*)(p + 8));
            }
#pragma unroll
            for (int g = 0; g < 4; ++g) {
                int n = (g << 9) + (wid << 4) + l16;
                const _Float16* p = Whh + (long)n * 512 + k0 + kbB;
                v16h bf = pack16(*(const v8h*)p, *(const v8h*)(p + 8));
                acc[0][g] = __builtin_amdgcn_wmma_f32_16x16x32_f16(
                    false, a[0], false, bf, (short)0, acc[0][g], false, false);
                acc[1][g] = __builtin_amdgcn_wmma_f32_16x16x32_f16(
                    false, a[1], false, bf, (short)0, acc[1][g], false, false);
            }
        }
        __syncthreads();  // all h_lds reads for step t complete

#pragma unroll
        for (int mi = 0; mi < 2; ++mi)
#pragma unroll
            for (int r = 0; r < 8; ++r) {
                int m = (mi << 4) + mbase + r;
                float ig = sigm(acc[mi][0][r]);
                float fg = sigm(acc[mi][1][r]);
                float gg = tanhf(acc[mi][2][r]);
                float og = sigm(acc[mi][3][r]);
                float cn = fg * c[mi][r] + ig * gg;
                c[mi][r] = cn;
                _Float16 hv = (_Float16)(og * tanhf(cn));
                h_lds[m][col] = hv;
                hout[((long)m * steps + ta) * ld_hout + col_off + col] = hv;
            }
        __syncthreads();  // h_lds updated for step t+1
    }
}

// ---------------------------------------------------------------------------
// Host-side GEMM launcher
// ---------------------------------------------------------------------------
template <bool BT, bool OUT16>
static inline void launch_gemm(hipStream_t st,
                               const _Float16* A, long lda, long sA1, long sA2,
                               const _Float16* Bm, long ldb, long sB1, long sB2,
                               void* C, long ldc, long sC1, long sC2,
                               const float* bias, const float* bias2,
                               int M, int N, int K, int z1n, int z2n) {
    dim3 g((N + 127) / 128, (M + 63) / 64, z1n * z2n);
    gemm_kernel<BT, OUT16><<<g, 256, 0, st>>>(A, lda, sA1, sA2, Bm, ldb, sB1, sB2,
                                              C, ldc, sC1, sC2, bias, bias2, M, N, K, z2n);
}

// ---------------------------------------------------------------------------
// kernel_launch
// ---------------------------------------------------------------------------
extern "C" void kernel_launch(void* const* d_in, const int* in_sizes, int n_in,
                              void* d_out, int out_size, void* d_ws, size_t ws_size,
                              hipStream_t stream) {
    (void)in_sizes; (void)n_in; (void)out_size; (void)ws_size;

    const int*   p_idx   = (const int*)  d_in[0];
    const float* p_mel   = (const float*)d_in[1];
    const float* p_emb   = (const float*)d_in[2];
    const float* p_wih_f = (const float*)d_in[3];
    const float* p_whh_f = (const float*)d_in[4];
    const float* p_bih_f = (const float*)d_in[5];
    const float* p_bhh_f = (const float*)d_in[6];
    const float* p_wih_b = (const float*)d_in[7];
    const float* p_whh_b = (const float*)d_in[8];
    const float* p_bih_b = (const float*)d_in[9];
    const float* p_bhh_b = (const float*)d_in[10];
    const float* p_wtp   = (const float*)d_in[11];
    const float* p_btp   = (const float*)d_in[12];
    const float* p_wmi   = (const float*)d_in[13];
    const float* p_bmi   = (const float*)d_in[14];
    const float* p_wai   = (const float*)d_in[15];
    const float* p_bai   = (const float*)d_in[16];
    const float* p_wao   = (const float*)d_in[17];
    const float* p_bao   = (const float*)d_in[18];
    const float* p_wdi   = (const float*)d_in[19];
    const float* p_wdh   = (const float*)d_in[20];
    const float* p_bdi   = (const float*)d_in[21];
    const float* p_bdh   = (const float*)d_in[22];
    const float* p_wmo   = (const float*)d_in[23];
    const float* p_bmo   = (const float*)d_in[24];

    char* ws = (char*)d_ws;
    size_t off = 0;
    auto alloc = [&](size_t bytes) -> size_t {
        size_t o = off; off += (bytes + 255) & ~(size_t)255; return o;
    };

    // f16 weights (converted each call; deterministic)
    size_t o_wihf = alloc(2048UL * 512 * 2);
    size_t o_whhf = alloc(2048UL * 512 * 2);
    size_t o_wihb = alloc(2048UL * 512 * 2);
    size_t o_whhb = alloc(2048UL * 512 * 2);
    size_t o_wtp  = alloc(512UL * 1024 * 2);
    size_t o_wmi  = alloc(512UL * 96 * 2);        // padded 80 -> 96
    size_t o_wai  = alloc(1536UL * 512 * 2);
    size_t o_wao  = alloc(512UL * 512 * 2);
    size_t o_wdi  = alloc(2048UL * 1024 * 2);
    size_t o_wdh  = alloc(2048UL * 512 * 2);
    size_t o_wmo  = alloc(80UL * 512 * 2);
    size_t o_bef  = alloc(2048UL * 4);            // bih+bhh fused biases
    size_t o_beb  = alloc(2048UL * 4);
    size_t o_bd   = alloc(2048UL * 4);

    // region A: enc gate precompute (2x134MB) -> scores+attn chunks -> dec gates
    size_t o_regA = alloc(2UL * 16384 * 2048 * 4);
    // region B: text_emb -> K buffer
    size_t o_regB = alloc(16384UL * 512 * 2);
    // region C: hs_concat -> ctx
    size_t o_regC = alloc(16384UL * 1024 * 2);
    size_t o_tenc = alloc(16384UL * 512 * 2);
    size_t o_msh  = alloc(32000UL * 96 * 2);
    size_t o_din  = alloc(32000UL * 1024 * 2);    // [mel_proj | attended]
    size_t o_regQ = alloc(32000UL * 512 * 2);     // Q -> dec_out
    size_t o_V    = alloc(16384UL * 512 * 2);

    _Float16* wihf = (_Float16*)(ws + o_wihf);
    _Float16* whhf = (_Float16*)(ws + o_whhf);
    _Float16* wihb = (_Float16*)(ws + o_wihb);
    _Float16* whhb = (_Float16*)(ws + o_whhb);
    _Float16* wtp  = (_Float16*)(ws + o_wtp);
    _Float16* wmi  = (_Float16*)(ws + o_wmi);
    _Float16* wai  = (_Float16*)(ws + o_wai);
    _Float16* wao  = (_Float16*)(ws + o_wao);
    _Float16* wdi  = (_Float16*)(ws + o_wdi);
    _Float16* wdh  = (_Float16*)(ws + o_wdh);
    _Float16* wmo  = (_Float16*)(ws + o_wmo);
    float* bef = (float*)(ws + o_bef);
    float* beb = (float*)(ws + o_beb);
    float* bd  = (float*)(ws + o_bd);

    float*    xg_f   = (float*)(ws + o_regA);
    float*    xg_b   = (float*)(ws + o_regA + 16384UL * 2048 * 4);
    float*    scores = (float*)(ws + o_regA);                          // reuse
    _Float16* attnp  = (_Float16*)(ws + o_regA + 256UL * 200 * 512 * 4);
    float*    xg_d   = (float*)(ws + o_regA);                          // reuse
    _Float16* temb   = (_Float16*)(ws + o_regB);
    _Float16* Kb     = (_Float16*)(ws + o_regB);                       // reuse
    _Float16* hs_cat = (_Float16*)(ws + o_regC);
    _Float16* ctx    = (_Float16*)(ws + o_regC);                       // reuse
    _Float16* tenc   = (_Float16*)(ws + o_tenc);
    _Float16* msh    = (_Float16*)(ws + o_msh);
    _Float16* din    = (_Float16*)(ws + o_din);
    _Float16* Qb     = (_Float16*)(ws + o_regQ);
    _Float16* dout16 = (_Float16*)(ws + o_regQ);                       // reuse
    _Float16* Vb     = (_Float16*)(ws + o_V);

    auto cvt = [&](const float* s, _Float16* d, long n) {
        cvt_kernel<<<dim3((unsigned)((n + 255) / 256)), 256, 0, stream>>>(s, d, n);
    };

    // 1) weight conversion + fused biases
    cvt(p_wih_f, wihf, 2048L * 512);
    cvt(p_whh_f, whhf, 2048L * 512);
    cvt(p_wih_b, wihb, 2048L * 512);
    cvt(p_whh_b, whhb, 2048L * 512);
    cvt(p_wtp,   wtp,  512L * 1024);
    cvt_pad_kernel<<<dim3((unsigned)((512L * 96 + 255) / 256)), 256, 0, stream>>>(p_wmi, wmi, 512, 80, 96);
    cvt(p_wai,   wai,  1536L * 512);
    cvt(p_wao,   wao,  512L * 512);
    cvt(p_wdi,   wdi,  2048L * 1024);
    cvt(p_wdh,   wdh,  2048L * 512);
    cvt(p_wmo,   wmo,  80L * 512);
    vecadd_kernel<<<8, 256, 0, stream>>>(p_bih_f, p_bhh_f, bef, 2048);
    vecadd_kernel<<<8, 256, 0, stream>>>(p_bih_b, p_bhh_b, beb, 2048);
    vecadd_kernel<<<8, 256, 0, stream>>>(p_bdi,   p_bdh,   bd,  2048);

    // 2) embedding gather
    embed_kernel<<<dim3(32768), 256, 0, stream>>>(p_idx, p_emb, temb, 16384);

    // 3) encoder gate precompute: xg = temb @ Wih^T + (bih+bhh)   [16384 x 2048]
    launch_gemm<true, false>(stream, temb, 512, 0, 0, wihf, 512, 0, 0,
                             xg_f, 2048, 0, 0, bef, nullptr, 16384, 2048, 512, 1, 1);
    launch_gemm<true, false>(stream, temb, 512, 0, 0, wihb, 512, 0, 0,
                             xg_b, 2048, 0, 0, beb, nullptr, 16384, 2048, 512, 1, 1);

    // 4) encoder recurrences (fwd into cols 0:512, bwd into cols 512:1024 of hs_cat)
    lstm_rec_kernel<<<1, 1024, 0, stream>>>(xg_f, whhf, hs_cat, 1024, 0,   512,  1);
    lstm_rec_kernel<<<1, 1024, 0, stream>>>(xg_b, whhb, hs_cat, 1024, 512, 512, -1);

    // 5) text projection: tenc = hs_cat @ tproj_W^T + b   [16384 x 512]
    launch_gemm<true, true>(stream, hs_cat, 1024, 0, 0, wtp, 1024, 0, 0,
                            tenc, 512, 0, 0, p_btp, nullptr, 16384, 512, 1024, 1, 1);

    // 6) shifted mel + input projection directly into dec_in[:, 0:512]
    mel_shift_kernel<<<dim3((unsigned)((32000L * 96 + 255) / 256)), 256, 0, stream>>>(p_mel, msh);
    launch_gemm<true, true>(stream, msh, 96, 0, 0, wmi, 96, 0, 0,
                            din, 1024, 0, 0, p_bmi, nullptr, 32000, 512, 96, 1, 1);

    // 7) attention projections (Q from mel_projected view of dec_in; K,V from tenc)
    launch_gemm<true, true>(stream, din, 1024, 0, 0, wai, 512, 0, 0,
                            Qb, 512, 0, 0, p_bai, nullptr, 32000, 512, 512, 1, 1);
    launch_gemm<true, true>(stream, tenc, 512, 0, 0, wai + 512L * 512, 512, 0, 0,
                            Kb, 512, 0, 0, p_bai + 512, nullptr, 16384, 512, 512, 1, 1);
    launch_gemm<true, true>(stream, tenc, 512, 0, 0, wai + 1024L * 512, 512, 0, 0,
                            Vb, 512, 0, 0, p_bai + 1024, nullptr, 16384, 512, 512, 1, 1);

    // 8) attention, chunked over T (5 x 200), batched over (b,h) = 32 x 8
    for (int ci = 0; ci < 5; ++ci) {
        long t0 = (long)ci * 200;
        // scores[b,h,tc,s] = Q[b,t0+tc,h,:] . K[b,s,h,:]
        launch_gemm<true, false>(stream,
            Qb + t0 * 512, 512, 1000L * 512, 64,
            Kb,            512, 512L * 512,  64,
            scores,        512, 8L * 200 * 512, 200L * 512,
            nullptr, nullptr, 200, 512, 64, 32, 8);
        softmax512_kernel<<<dim3(32 * 8 * 200), 256, 0, stream>>>(scores, attnp);
        // ctx[b,t0+tc,h,:] = attn[b,h,tc,:] @ V[b,:,h,:]
        launch_gemm<false, true>(stream,
            attnp,         512, 8L * 200 * 512, 200L * 512,
            Vb,            512, 512L * 512,  64,
            ctx + t0 * 512, 512, 1000L * 512, 64,
            nullptr, nullptr, 200, 64, 512, 32, 8);
    }

    // 9) attention out-proj directly into dec_in[:, 512:1024]
    launch_gemm<true, true>(stream, ctx, 512, 0, 0, wao, 512, 0, 0,
                            din + 512, 1024, 0, 0, p_bao, nullptr, 32000, 512, 512, 1, 1);

    // 10) decoder gate precompute: xg_d = dec_in @ dec_Wih^T + (bih+bhh) [32000 x 2048]
    launch_gemm<true, false>(stream, din, 1024, 0, 0, wdi, 1024, 0, 0,
                             xg_d, 2048, 0, 0, bd, nullptr, 32000, 2048, 1024, 1, 1);

    // 11) decoder recurrence
    lstm_rec_kernel<<<1, 1024, 0, stream>>>(xg_d, wdh, dout16, 512, 0, 1000, 1);

    // 12) output projection -> d_out f32 [32000 x 80]
    launch_gemm<true, false>(stream, dout16, 512, 0, 0, wmo, 512, 0, 0,
                             d_out, 80, 0, 0, p_bmo, nullptr, 32000, 80, 512, 1, 1);
}
